// CoAttLayer_49598282334520
// MI455X (gfx1250) — compile-verified, matrix operand order checked by reading, not run on hardware
//
#include <hip/hip_runtime.h>

typedef __attribute__((ext_vector_type(16))) __bf16 v16bf;
typedef __attribute__((ext_vector_type(8)))  float  v8f;
typedef __attribute__((ext_vector_type(4)))  unsigned int u32x4;
typedef __attribute__((ext_vector_type(8)))  int i32x8;
typedef __attribute__((ext_vector_type(4)))  int i32x4;

#define BB 64
#define TT 1024
#define NN 512
#define DD 64
#define KK 80
#define KP 96

#if defined(__has_builtin)
# if __has_builtin(__builtin_amdgcn_tensor_load_to_lds) && __has_builtin(__builtin_amdgcn_s_wait_tensorcnt)
#  define USE_TDM 1
# endif
#endif
#ifndef USE_TDM
# define USE_TDM 0
#endif

// Per-lane fragment gather matching the CDNA5 16-bit A/B VGPR layout:
// lane l: row = (l&15), elements kb..kb+7 and kb+16..kb+23 with kb=(l>>4)*8.
union FragU { uint4 u[2]; v16bf v; };
__device__ __forceinline__ v16bf ldfrag(const __bf16* p) {
    FragU f;
    f.u[0] = *(const uint4*)(p);
    f.u[1] = *(const uint4*)(p + 16);
    return f.v;
}

// Cooperative fallback tile copy (also used for the one-shot tiles)
__device__ __forceinline__ void copy_tile(__bf16* lds, const __bf16* g,
                                          int cols, int rows, int gstride,
                                          int tid, int nthreads) {
    const int segs = cols >> 3;
    for (int i = tid; i < rows * segs; i += nthreads) {
        int r = i / segs, s = i % segs;
        *(uint4*)(lds + r * cols + s * 8) =
            *(const uint4*)(g + (size_t)r * gstride + s * 8);
    }
}

#if USE_TDM
// Issue one TDM 2-D tile load (tile1 rows x tile0 bf16 cols) into LDS.
// D# bit packing per CDNA5 ISA.
__device__ __forceinline__ void tdm_load_2d(void* lds, const void* gptr,
                                            unsigned tile0, unsigned tile1,
                                            unsigned tdim0, unsigned tdim1,
                                            unsigned stride0) {
    unsigned long long ga = (unsigned long long)(size_t)gptr;
    unsigned ldsoff = (unsigned)(size_t)lds;
    u32x4 g0;
    g0[0] = 1u;                                   // count=1, user desc
    g0[1] = ldsoff;                               // lds_addr
    g0[2] = (unsigned)ga;                         // global_addr[31:0]
    g0[3] = (unsigned)((ga >> 32) & 0x01FFFFFFu) | (2u << 30);  // ga[56:32], type=2
    i32x8 g1;
    g1[0] = (int)(1u << 16);                      // data_size=1 (2 bytes)
    g1[1] = (int)((tdim0 & 0xFFFFu) << 16);       // tensor_dim0[15:0]
    g1[2] = (int)((tdim0 >> 16) | ((tdim1 & 0xFFFFu) << 16));
    g1[3] = (int)((tdim1 >> 16) | (tile0 << 16)); // tile_dim0
    g1[4] = (int)(tile1 & 0xFFFFu);               // tile_dim1 (tile_dim2=0)
    g1[5] = (int)stride0;                         // tensor_dim0_stride[31:0]
    g1[6] = 0;                                    // stride0[47:32], stride1 lo
    g1[7] = 0;
    i32x4 z4 = {0, 0, 0, 0};
#if __clang_major__ >= 23
    i32x8 z8 = {0, 0, 0, 0, 0, 0, 0, 0};
    __builtin_amdgcn_tensor_load_to_lds(g0, g1, z4, z4, z8, 0);
#else
    __builtin_amdgcn_tensor_load_to_lds(g0, g1, z4, z4, 0);
#endif
}
#endif

// ---------------------------------------------------------------------------
// Kernel 0: convert inputs to bf16, transpose Wl, zero-pad K 80->96
// ---------------------------------------------------------------------------
__global__ __launch_bounds__(256) void cvt_kernel(
    const float* __restrict__ review, const float* __restrict__ post,
    const float* __restrict__ Wl, const float* __restrict__ Wr,
    const float* __restrict__ Wp, const float* __restrict__ whr,
    const float* __restrict__ whp,
    __bf16* __restrict__ review_bf, __bf16* __restrict__ post_bf,
    __bf16* __restrict__ wlt, __bf16* __restrict__ wrp, __bf16* __restrict__ wpp,
    float* __restrict__ whr_pad, float* __restrict__ whp_pad)
{
    const long n1 = (long)BB * TT * DD;
    const long n2 = (long)BB * NN * DD;
    const long n3 = 64L * 64;
    const long n4 = (long)KP * 64;
    const long n5 = (long)KP * 64;
    const long n6 = KP;
    const long n7 = KP;
    const long tot = n1 + n2 + n3 + n4 + n5 + n6 + n7;
    for (long i = (long)blockIdx.x * blockDim.x + threadIdx.x; i < tot;
         i += (long)gridDim.x * blockDim.x) {
        long j = i;
        if (j < n1) { review_bf[j] = (__bf16)review[j]; continue; }
        j -= n1;
        if (j < n2) { post_bf[j] = (__bf16)post[j]; continue; }
        j -= n2;
        if (j < n3) { int dp = (int)(j >> 6), d2 = (int)(j & 63);
                      wlt[j] = (__bf16)Wl[d2 * 64 + dp]; continue; }
        j -= n3;
        if (j < n4) { int k = (int)(j >> 6), d2 = (int)(j & 63);
                      wrp[j] = (__bf16)((k < KK) ? Wr[k * 64 + d2] : 0.f); continue; }
        j -= n4;
        if (j < n5) { int k = (int)(j >> 6), d2 = (int)(j & 63);
                      wpp[j] = (__bf16)((k < KK) ? Wp[k * 64 + d2] : 0.f); continue; }
        j -= n5;
        if (j < n6) { whp_pad[j] = (j < KK) ? whp[j] : 0.f; continue; }
        j -= n6;
        whr_pad[j] = (j < KK) ? whr[j] : 0.f;
    }
}

// ---------------------------------------------------------------------------
// Kernel 1: generic C[M,Nc] = A[M,64] * Bm[Nc,64]^T  (bf16 in/out)
// ---------------------------------------------------------------------------
__global__ __launch_bounds__(32) void gemm_abT(
    const __bf16* __restrict__ A, const __bf16* __restrict__ Bm,
    __bf16* __restrict__ C, int Nc, long strideA, long strideB, long strideC)
{
    const int bz = blockIdx.z;
    A  += (size_t)bz * strideA;
    Bm += (size_t)bz * strideB;
    C  += (size_t)bz * strideC;
    const int lane = threadIdx.x & 31;
    const int l16  = lane & 15;
    const int kb   = (lane >> 4) * 8;
    const int m    = blockIdx.y * 16 + l16;
    const int n    = blockIdx.x * 16 + l16;
    v8f acc = {};
#pragma unroll
    for (int ks = 0; ks < 64; ks += 32) {
        v16bf a = ldfrag(A  + (size_t)m * 64 + ks + kb);
        v16bf b = ldfrag(Bm + (size_t)n * 64 + ks + kb);
        acc = __builtin_amdgcn_wmma_f32_16x16x32_bf16(false, a, false, b,
                                                      (short)0, acc, false, false);
    }
#pragma unroll
    for (int r = 0; r < 8; ++r) {
        int mr = blockIdx.y * 16 + r + (lane >> 4) * 8;
        C[(size_t)mr * Nc + n] = (__bf16)acc[r];
    }
}

// ---------------------------------------------------------------------------
// Kernel 2: fused Hp path. Grid (N/64, B).
//   t-loop: L = tanh(Rl*post^T) (WMMA) -> LDS as L^T; M[k,n] += Rr*L (WMMA)
//   Rl/Rr tiles streamed by TDM with double buffering.
// ---------------------------------------------------------------------------
__global__ __launch_bounds__(256) void hp_kernel(
    const __bf16* __restrict__ rl,    // [B,T,64]
    const __bf16* __restrict__ postb, // [B,N,64]
    const __bf16* __restrict__ rr,    // [B,KP,T]
    const __bf16* __restrict__ pp,    // [B,KP,N]
    const float*  __restrict__ whp,   // [KP]
    float* __restrict__ logits_p)     // [B,N]
{
    __shared__ __align__(16) __bf16 s_post[64 * 64];
    __shared__ __align__(16) __bf16 s_rl[2][64 * 64];
    __shared__ __align__(16) __bf16 s_rr[2][KP * 64];
    __shared__ __align__(16) __bf16 s_lt[64 * 64];   // L^T: [n][t]
    __shared__ float s_part[8][64];

    const int b    = blockIdx.y;
    const int n0   = blockIdx.x * 64;
    const int tid  = threadIdx.x;
    const int wv   = tid >> 5;
    const int lane = tid & 31;
    const int l16  = lane & 15;
    const int kb   = (lane >> 4) * 8;

    s_part[wv][lane] = 0.f;
    s_part[wv][lane + 32] = 0.f;

    copy_tile(s_post, postb + (size_t)b * NN * DD + (size_t)n0 * DD,
              64, 64, 64, tid, 256);

    const __bf16* rl_b = rl + (size_t)b * TT * DD;
    const __bf16* rr_b = rr + (size_t)b * KP * TT;

    auto issue = [&](int bufi, int tii) {
#if USE_TDM
        if (wv == 0) {
            tdm_load_2d(&s_rl[bufi][0], rl_b + (size_t)tii * 64 * DD,
                        64, 64, 64, 64, 64);
            tdm_load_2d(&s_rr[bufi][0], rr_b + tii * 64,
                        64, KP, TT, KP, TT);
        }
#else
        copy_tile(&s_rl[bufi][0], rl_b + (size_t)tii * 64 * DD, 64, 64, 64, tid, 256);
        copy_tile(&s_rr[bufi][0], rr_b + tii * 64, 64, KP, TT, tid, 256);
#endif
    };

    v8f macc[3] = {};
    issue(0, 0);

    for (int ti = 0; ti < TT / 64; ++ti) {
        const bool pref = (ti + 1) < (TT / 64);
        if (pref) issue((ti + 1) & 1, ti + 1);
#if USE_TDM
        if (wv == 0) {
            if (pref) __builtin_amdgcn_s_wait_tensorcnt(2);
            else      __builtin_amdgcn_s_wait_tensorcnt(0);
        }
#endif
        __syncthreads();   // tile ti visible to all waves
        const __bf16* crl = &s_rl[ti & 1][0];
        const __bf16* crr = &s_rr[ti & 1][0];

        // L subtiles: 16 of them, wave w does {w, w+8}
#pragma unroll
        for (int rep = 0; rep < 2; ++rep) {
            int idx = wv + rep * 8;
            int tt = idx >> 2, nt = idx & 3;
            v8f acc = {};
#pragma unroll
            for (int ks = 0; ks < 64; ks += 32) {
                v16bf a  = ldfrag(&crl[(tt * 16 + l16) * 64 + ks + kb]);
                v16bf bf = ldfrag(&s_post[(nt * 16 + l16) * 64 + ks + kb]);
                acc = __builtin_amdgcn_wmma_f32_16x16x32_bf16(false, a, false, bf,
                                                              (short)0, acc, false, false);
            }
            int n = nt * 16 + l16;
#pragma unroll
            for (int r = 0; r < 8; ++r) {
                int t = tt * 16 + r + kb;
                s_lt[n * 64 + t] = (__bf16)tanhf(acc[r]);
            }
        }
        __syncthreads();

        // M[k,n] += Rr(k,t) * L(t,n)
#pragma unroll
        for (int s = 0; s < 3; ++s) {
            int idx = wv * 3 + s;
            int kt = idx >> 2, nt = idx & 3;
#pragma unroll
            for (int ks = 0; ks < 64; ks += 32) {
                v16bf a  = ldfrag(&crr[(kt * 16 + l16) * 64 + ks + kb]);
                v16bf bf = ldfrag(&s_lt[(nt * 16 + l16) * 64 + ks + kb]);
                macc[s] = __builtin_amdgcn_wmma_f32_16x16x32_bf16(false, a, false, bf,
                                                                  (short)0, macc[s], false, false);
            }
        }
        __syncthreads();   // protect buffer reuse by next issue
    }

    // epilogue: Hp = tanh(Pp + M); deterministic logit reduction
#pragma unroll
    for (int s = 0; s < 3; ++s) {
        int idx = wv * 3 + s;
        int kt = idx >> 2, nt = idx & 3;
        int n = n0 + nt * 16 + l16;
        float part = 0.f;
#pragma unroll
        for (int r = 0; r < 8; ++r) {
            int k = kt * 16 + r + kb;
            float ppv = (float)pp[(size_t)b * KP * NN + (size_t)k * NN + n];
            part += whp[k] * tanhf(ppv + macc[s][r]);
        }
        part += __shfl_down(part, 16);
        if (lane < 16) s_part[wv][nt * 16 + l16] += part;
    }
    __syncthreads();
    if (tid < 64) {
        float v = 0.f;
#pragma unroll
        for (int w = 0; w < 8; ++w) v += s_part[w][tid];
        logits_p[(size_t)b * NN + n0 + tid] = v;
    }
}

// ---------------------------------------------------------------------------
// Kernel 3: fused Hr path. Grid (T/64, B).
//   n-loop: recompute L tile -> LDS [t][n]; Q[k,t] += Pp * L^T (WMMA)
//   post/Pp tiles streamed by TDM with double buffering.
// ---------------------------------------------------------------------------
__global__ __launch_bounds__(256) void hr_kernel(
    const __bf16* __restrict__ rl,
    const __bf16* __restrict__ postb,
    const __bf16* __restrict__ rr,
    const __bf16* __restrict__ pp,
    const float*  __restrict__ whr,
    float* __restrict__ logits_r)     // [B,T]
{
    __shared__ __align__(16) __bf16 s_rl[64 * 64];
    __shared__ __align__(16) __bf16 s_post[2][64 * 64];
    __shared__ __align__(16) __bf16 s_pp[2][KP * 64];
    __shared__ __align__(16) __bf16 s_l[64 * 64];    // L: [t][n]
    __shared__ float s_part[8][64];

    const int b    = blockIdx.y;
    const int t0   = blockIdx.x * 64;
    const int tid  = threadIdx.x;
    const int wv   = tid >> 5;
    const int lane = tid & 31;
    const int l16  = lane & 15;
    const int kb   = (lane >> 4) * 8;

    s_part[wv][lane] = 0.f;
    s_part[wv][lane + 32] = 0.f;

    copy_tile(s_rl, rl + (size_t)b * TT * DD + (size_t)t0 * DD,
              64, 64, 64, tid, 256);

    const __bf16* po_b = postb + (size_t)b * NN * DD;
    const __bf16* pp_b = pp + (size_t)b * KP * NN;

    auto issue = [&](int bufi, int nii) {
#if USE_TDM
        if (wv == 0) {
            tdm_load_2d(&s_post[bufi][0], po_b + (size_t)nii * 64 * DD,
                        64, 64, 64, 64, 64);
            tdm_load_2d(&s_pp[bufi][0], pp_b + nii * 64,
                        64, KP, NN, KP, NN);
        }
#else
        copy_tile(&s_post[bufi][0], po_b + (size_t)nii * 64 * DD, 64, 64, 64, tid, 256);
        copy_tile(&s_pp[bufi][0], pp_b + nii * 64, 64, KP, NN, tid, 256);
#endif
    };

    v8f qacc[3] = {};
    issue(0, 0);

    for (int ni = 0; ni < NN / 64; ++ni) {
        const bool pref = (ni + 1) < (NN / 64);
        if (pref) issue((ni + 1) & 1, ni + 1);
#if USE_TDM
        if (wv == 0) {
            if (pref) __builtin_amdgcn_s_wait_tensorcnt(2);
            else      __builtin_amdgcn_s_wait_tensorcnt(0);
        }
#endif
        __syncthreads();
        const __bf16* cpo = &s_post[ni & 1][0];
        const __bf16* cpp = &s_pp[ni & 1][0];

        // L subtiles, stored natural orientation [t][n]
#pragma unroll
        for (int rep = 0; rep < 2; ++rep) {
            int idx = wv + rep * 8;
            int tt = idx >> 2, nt = idx & 3;
            v8f acc = {};
#pragma unroll
            for (int ks = 0; ks < 64; ks += 32) {
                v16bf a  = ldfrag(&s_rl[(tt * 16 + l16) * 64 + ks + kb]);
                v16bf bf = ldfrag(&cpo[(nt * 16 + l16) * 64 + ks + kb]);
                acc = __builtin_amdgcn_wmma_f32_16x16x32_bf16(false, a, false, bf,
                                                              (short)0, acc, false, false);
            }
            int n = nt * 16 + l16;
#pragma unroll
            for (int r = 0; r < 8; ++r) {
                int t = tt * 16 + r + kb;
                s_l[t * 64 + n] = (__bf16)tanhf(acc[r]);
            }
        }
        __syncthreads();

        // Q[k,t] += Pp(k,n) * L(t,n)^T
#pragma unroll
        for (int s = 0; s < 3; ++s) {
            int idx = wv * 3 + s;
            int kt = idx >> 2, tt = idx & 3;
#pragma unroll
            for (int ks = 0; ks < 64; ks += 32) {
                v16bf a  = ldfrag(&cpp[(kt * 16 + l16) * 64 + ks + kb]);
                v16bf bf = ldfrag(&s_l[(tt * 16 + l16) * 64 + ks + kb]);
                qacc[s] = __builtin_amdgcn_wmma_f32_16x16x32_bf16(false, a, false, bf,
                                                                  (short)0, qacc[s], false, false);
            }
        }
        __syncthreads();
    }

#pragma unroll
    for (int s = 0; s < 3; ++s) {
        int idx = wv * 3 + s;
        int kt = idx >> 2, tt = idx & 3;
        int t = t0 + tt * 16 + l16;
        float part = 0.f;
#pragma unroll
        for (int r = 0; r < 8; ++r) {
            int k = kt * 16 + r + kb;
            float rrv = (float)rr[(size_t)b * KP * TT + (size_t)k * TT + t];
            part += whr[k] * tanhf(rrv + qacc[s][r]);
        }
        part += __shfl_down(part, 16);
        if (lane < 16) s_part[wv][tt * 16 + l16] += part;
    }
    __syncthreads();
    if (tid < 64) {
        float v = 0.f;
#pragma unroll
        for (int w = 0; w < 8; ++w) v += s_part[w][tid];
        logits_r[(size_t)b * TT + t0 + tid] = v;
    }
}

// ---------------------------------------------------------------------------
// Kernel 4: softmax + attention pooling, one block per batch
// ---------------------------------------------------------------------------
__global__ __launch_bounds__(256) void finalize_kernel(
    const float* __restrict__ logits_p, const float* __restrict__ logits_r,
    const float* __restrict__ postf, const float* __restrict__ reviewf,
    float* __restrict__ out)
{
    __shared__ float sh[1024];
    __shared__ float red[256];
    const int b = blockIdx.x;
    const int tid = threadIdx.x;
    const int d = tid & 63, g = tid >> 6;

    {
        float l0 = logits_p[(size_t)b * NN + tid];
        float l1 = logits_p[(size_t)b * NN + 256 + tid];
        red[tid] = fmaxf(l0, l1);
        __syncthreads();
        for (int s = 128; s > 0; s >>= 1) {
            if (tid < s) red[tid] = fmaxf(red[tid], red[tid + s]);
            __syncthreads();
        }
        float mx = red[0];
        __syncthreads();
        float e0 = expf(l0 - mx), e1 = expf(l1 - mx);
        sh[tid] = e0; sh[256 + tid] = e1;
        red[tid] = e0 + e1;
        __syncthreads();
        for (int s = 128; s > 0; s >>= 1) {
            if (tid < s) red[tid] += red[tid + s];
            __syncthreads();
        }
        float sum = red[0];
        __syncthreads();
        float acc = 0.f;
        for (int n = g; n < NN; n += 4)
            acc += postf[(size_t)b * NN * DD + (size_t)n * DD + d] * sh[n];
        red[tid] = acc;
        __syncthreads();
        if (tid < 64)
            out[(size_t)b * 128 + tid] =
                (red[tid] + red[tid + 64] + red[tid + 128] + red[tid + 192]) / sum;
        __syncthreads();
    }

    {
        float l[4], m = -3.4e38f;
#pragma unroll
        for (int j = 0; j < 4; ++j) {
            l[j] = logits_r[(size_t)b * TT + j * 256 + tid];
            m = fmaxf(m, l[j]);
        }
        red[tid] = m;
        __syncthreads();
        for (int s = 128; s > 0; s >>= 1) {
            if (tid < s) red[tid] = fmaxf(red[tid], red[tid + s]);
            __syncthreads();
        }
        float mx = red[0];
        __syncthreads();
        float psum = 0.f;
#pragma unroll
        for (int j = 0; j < 4; ++j) {
            float e = expf(l[j] - mx);
            sh[j * 256 + tid] = e;
            psum += e;
        }
        red[tid] = psum;
        __syncthreads();
        for (int s = 128; s > 0; s >>= 1) {
            if (tid < s) red[tid] += red[tid + s];
            __syncthreads();
        }
        float sum = red[0];
        __syncthreads();
        float acc = 0.f;
        for (int t = g; t < TT; t += 4)
            acc += reviewf[(size_t)b * TT * DD + (size_t)t * DD + d] * sh[t];
        red[tid] = acc;
        __syncthreads();
        if (tid < 64)
            out[(size_t)b * 128 + 64 + tid] =
                (red[tid] + red[tid + 64] + red[tid + 128] + red[tid + 192]) / sum;
    }
}

// ---------------------------------------------------------------------------
extern "C" void kernel_launch(void* const* d_in, const int* in_sizes, int n_in,
                              void* d_out, int out_size, void* d_ws, size_t ws_size,
                              hipStream_t stream) {
    (void)in_sizes; (void)n_in; (void)out_size; (void)ws_size;
    const float* review = (const float*)d_in[0];
    const float* post   = (const float*)d_in[1];
    const float* Wl     = (const float*)d_in[2];
    const float* Wr     = (const float*)d_in[3];
    const float* Wp     = (const float*)d_in[4];
    const float* whr    = (const float*)d_in[5];
    const float* whp    = (const float*)d_in[6];
    float* out = (float*)d_out;

    char* ws = (char*)d_ws;
    size_t off = 0;
    auto take = [&](size_t bytes) -> char* {
        off = (off + 255) & ~(size_t)255;
        char* p = ws + off;
        off += bytes;
        return p;
    };
    __bf16* review_bf = (__bf16*)take((size_t)BB * TT * DD * 2);
    __bf16* post_bf   = (__bf16*)take((size_t)BB * NN * DD * 2);
    __bf16* wlt       = (__bf16*)take((size_t)64 * 64 * 2);
    __bf16* wrp       = (__bf16*)take((size_t)KP * 64 * 2);
    __bf16* wpp       = (__bf16*)take((size_t)KP * 64 * 2);
    float*  whp_pad   = (float*)take((size_t)KP * 4);
    float*  whr_pad   = (float*)take((size_t)KP * 4);
    __bf16* rl_bf     = (__bf16*)take((size_t)BB * TT * DD * 2);
    __bf16* rr_bf     = (__bf16*)take((size_t)BB * KP * TT * 2);
    __bf16* pp_bf     = (__bf16*)take((size_t)BB * KP * NN * 2);
    float*  logits_p  = (float*)take((size_t)BB * NN * 4);
    float*  logits_r  = (float*)take((size_t)BB * TT * 4);

    cvt_kernel<<<dim3(2048), dim3(256), 0, stream>>>(
        review, post, Wl, Wr, Wp, whr, whp,
        review_bf, post_bf, wlt, wrp, wpp, whr_pad, whp_pad);

    gemm_abT<<<dim3(64 / 16, (BB * TT) / 16, 1), 32, 0, stream>>>(
        review_bf, wlt, rl_bf, 64, 0, 0, 0);
    gemm_abT<<<dim3(TT / 16, KP / 16, BB), 32, 0, stream>>>(
        wrp, review_bf, rr_bf, TT, 0, (long)TT * DD, (long)KP * TT);
    gemm_abT<<<dim3(NN / 16, KP / 16, BB), 32, 0, stream>>>(
        wpp, post_bf, pp_bf, NN, 0, (long)NN * DD, (long)KP * NN);

    hp_kernel<<<dim3(NN / 64, BB), 256, 0, stream>>>(
        rl_bf, post_bf, rr_bf, pp_bf, whp_pad, logits_p);
    hr_kernel<<<dim3(TT / 64, BB), 256, 0, stream>>>(
        rl_bf, post_bf, rr_bf, pp_bf, whr_pad, logits_r);

    finalize_kernel<<<dim3(BB), 256, 0, stream>>>(
        logits_p, logits_r, post, review, out);
}
